// CrossAttention_37357625541201
// MI455X (gfx1250) — matrix operand order repaired and checked by reading, not compiled, not run on hardware
//
#include <hip/hip_runtime.h>
#include <hip/hip_bf16.h>

#define NB 8
#define QL 256
#define KL 64
#define SS 64
#define DD 128
#define HH 8
#define LS (KL * SS)   /* 4096 */
#define HD (HH * DD)   /* 1024 */

typedef __bf16 bf16_t;
typedef __attribute__((ext_vector_type(16))) __bf16 v16bf;
typedef __attribute__((ext_vector_type(8)))  __bf16 v8bf;
typedef __attribute__((ext_vector_type(4)))  __bf16 v4bf;
typedef __attribute__((ext_vector_type(8)))  float  v8f;

// ---------------- WMMA fragment helpers (wave32, ISA 7.12.2 layouts) ----------------
// A 16x32 bf16: lane m = lane&15, kh = lane>>4; elems 0..7 -> K = k0+kh*8+e,
//               elems 8..15 -> K = k0+16+kh*8+(e-8).
__device__ __forceinline__ v16bf frag_a_bf16(const bf16_t* __restrict__ row, int k0, int lane) {
  const bf16_t* p = row + k0 + ((lane >> 4) << 3);
  v8bf lo = *(const v8bf*)(p);
  v8bf hi = *(const v8bf*)(p + 16);
  v16bf f;
#pragma unroll
  for (int i = 0; i < 8; ++i) { f[i] = lo[i]; f[i + 8] = hi[i]; }
  return f;
}
// B 32x16 bf16: lane n = lane&15, kh = lane>>4; elems 0..7 -> K = k0+kh*8+e,
// elems 8..15 -> K = k0+16+kh*8+(e-8) (B column n supplied as a contiguous-K row, i.e. B^T row-major).
__device__ __forceinline__ v16bf frag_b_bf16(const bf16_t* __restrict__ colrow, int k0, int lane) {
  const bf16_t* p = colrow + k0 + ((lane >> 4) << 3);
  v8bf lo = *(const v8bf*)(p);
  v8bf hi = *(const v8bf*)(p + 16);
  v16bf f;
#pragma unroll
  for (int i = 0; i < 8; ++i) { f[i] = lo[i]; f[i + 8] = hi[i]; }
  return f;
}
__device__ __forceinline__ v8f wmma_bf16(v16bf a, v16bf b, v8f c) {
  return __builtin_amdgcn_wmma_f32_16x16x32_bf16(false, a, false, b, (short)0, c, false, false);
}
// C/D 16x16 f32: reg r, lane l -> element (m = r + 8*(l>>4), n = l&15)
__device__ __forceinline__ void store_bf16(bf16_t* __restrict__ C, int ldc, int row0, int col0,
                                           v8f acc, int lane) {
  int n = lane & 15, mh = (lane >> 4) << 3;
#pragma unroll
  for (int r = 0; r < 8; ++r)
    C[(size_t)(row0 + mh + r) * ldc + col0 + n] = (__bf16)acc[r];
}
__device__ __forceinline__ void store_f32(float* __restrict__ C, int ldc, int row0, int col0,
                                          v8f acc, int lane, float scale) {
  int n = lane & 15, mh = (lane >> 4) << 3;
#pragma unroll
  for (int r = 0; r < 8; ++r)
    C[(size_t)(row0 + mh + r) * ldc + col0 + n] = acc[r] * scale;
}

// ---------------- Kernels ----------------

// Vectorized f32 -> bf16 conversion (4 elements / thread).
__global__ void cvt_bf16_kernel(const float* __restrict__ src, bf16_t* __restrict__ dst, int n4) {
  int i = blockIdx.x * blockDim.x + threadIdx.x;
  if (i >= n4) return;
  float4 x = ((const float4*)src)[i];
  v4bf y = { (__bf16)x.x, (__bf16)x.y, (__bf16)x.z, (__bf16)x.w };
  ((v4bf*)dst)[i] = y;
}

// C_bf16[M,128] = A_bf16[M,128] @ W_bf16[128,128]^T  (q/k/v projections, K = D = 128)
__global__ void proj_gemm_kernel(const bf16_t* __restrict__ A, const bf16_t* __restrict__ W,
                                 bf16_t* __restrict__ C, int M) {
  int lane = threadIdx.x & 31;
  int gw = blockIdx.x * (blockDim.x >> 5) + (threadIdx.x >> 5);
  const int tiles_n = DD / 32;
  if (gw >= (M / 32) * tiles_n) return;
  int row0 = (gw / tiles_n) * 32, col0 = (gw % tiles_n) * 32;
  const bf16_t* arow = A + (size_t)(row0 + (lane & 15)) * DD;
  const bf16_t* brow = W + (size_t)(col0 + (lane & 15)) * DD;
  v8f c00 = {}, c01 = {}, c10 = {}, c11 = {};
#pragma unroll 1
  for (int k = 0; k < DD; k += 32) {
    v16bf a0 = frag_a_bf16(arow, k, lane);
    v16bf a1 = frag_a_bf16(arow + 16 * DD, k, lane);
    v16bf b0 = frag_b_bf16(brow, k, lane);
    v16bf b1 = frag_b_bf16(brow + 16 * DD, k, lane);
    c00 = wmma_bf16(a0, b0, c00);
    c01 = wmma_bf16(a0, b1, c01);
    c10 = wmma_bf16(a1, b0, c10);
    c11 = wmma_bf16(a1, b1, c11);
  }
  store_bf16(C, DD, row0, col0, c00, lane);
  store_bf16(C, DD, row0, col0 + 16, c01, lane);
  store_bf16(C, DD, row0 + 16, col0, c10, lane);
  store_bf16(C, DD, row0 + 16, col0 + 16, c11, lane);
}

// E[QL,LS] = (Q[QL,D] @ K[LS,D]^T) / sqrt(D)   (one batch)
__global__ void energy_gemm_kernel(const bf16_t* __restrict__ Q, const bf16_t* __restrict__ Kp,
                                   float* __restrict__ E) {
  int lane = threadIdx.x & 31;
  int gw = blockIdx.x * (blockDim.x >> 5) + (threadIdx.x >> 5);
  const int tiles_n = LS / 32;
  if (gw >= (QL / 32) * tiles_n) return;
  int row0 = (gw / tiles_n) * 32, col0 = (gw % tiles_n) * 32;
  const bf16_t* arow = Q + (size_t)(row0 + (lane & 15)) * DD;
  const bf16_t* brow = Kp + (size_t)(col0 + (lane & 15)) * DD;
  v8f c00 = {}, c01 = {}, c10 = {}, c11 = {};
#pragma unroll 1
  for (int k = 0; k < DD; k += 32) {
    v16bf a0 = frag_a_bf16(arow, k, lane);
    v16bf a1 = frag_a_bf16(arow + 16 * DD, k, lane);
    v16bf b0 = frag_b_bf16(brow, k, lane);
    v16bf b1 = frag_b_bf16(brow + 16 * DD, k, lane);
    c00 = wmma_bf16(a0, b0, c00);
    c01 = wmma_bf16(a0, b1, c01);
    c10 = wmma_bf16(a1, b0, c10);
    c11 = wmma_bf16(a1, b1, c11);
  }
  const float s = 0.08838834764831845f;  // 1/sqrt(128)
  store_f32(E, LS, row0, col0, c00, lane, s);
  store_f32(E, LS, row0, col0 + 16, c01, lane, s);
  store_f32(E, LS, row0 + 16, col0, c10, lane, s);
  store_f32(E, LS, row0 + 16, col0 + 16, c11, lane, s);
}

// Masked softmax over KL for one batch. One thread per (q,s): the 64 head-
// independent energies live in registers; the 256 MB mask stream is read exactly
// once and bit-packed into a u64 per head.
__global__ void softmax_kernel(const float* __restrict__ E, const int* __restrict__ maskn,
                               bf16_t* __restrict__ attn) {
  int idx = blockIdx.x * blockDim.x + threadIdx.x;
  if (idx >= QL * SS) return;
  int q = idx / SS, s = idx % SS;
  float ev[KL];
  const float* erow = E + (size_t)q * LS + s;
#pragma unroll
  for (int l = 0; l < KL; ++l) ev[l] = erow[l * SS];
  for (int h = 0; h < HH; ++h) {
    const int* mrow = maskn + ((size_t)(h * QL + q) * KL) * SS + s;
    unsigned long long bits = 0ull;
    float mx = -3.0e38f;
#pragma unroll
    for (int l = 0; l < KL; ++l) {
      bool keep = (mrow[(size_t)l * SS] != 0);
      bits |= ((unsigned long long)(keep ? 1u : 0u)) << l;
      float v = keep ? ev[l] : -1.0e20f;
      mx = fmaxf(mx, v);
    }
    float sum = 0.f;
#pragma unroll
    for (int l = 0; l < KL; ++l) {
      float v = ((bits >> l) & 1ull) ? ev[l] : -1.0e20f;
      sum += __expf(v - mx);
    }
    float rs = 1.0f / sum;
    bf16_t* orow = attn + (size_t)(h * QL + q) * LS + s;
#pragma unroll
    for (int l = 0; l < KL; ++l) {
      float v = ((bits >> l) & 1ull) ? ev[l] : -1.0e20f;
      orow[(size_t)l * SS] = (__bf16)(__expf(v - mx) * rs);
    }
  }
}

// vT[n][d][ls] = v[n][ls][d]  (LDS 32x32 tile transpose, all batches)
__global__ void transpose_v_kernel(const bf16_t* __restrict__ vb, bf16_t* __restrict__ vT) {
  __shared__ bf16_t tile[32][33];
  int n = blockIdx.z, ls0 = blockIdx.x * 32, d0 = blockIdx.y * 32;
  const bf16_t* src = vb + (size_t)n * LS * DD;
  bf16_t* dst = vT + (size_t)n * DD * LS;
  for (int i = threadIdx.y; i < 32; i += 8)
    tile[i][threadIdx.x] = src[(size_t)(ls0 + i) * DD + d0 + threadIdx.x];
  __syncthreads();
  for (int i = threadIdx.y; i < 32; i += 8)
    dst[(size_t)(d0 + i) * LS + ls0 + threadIdx.x] = tile[threadIdx.x][i];
}

// ctx[q, h*128+d] = sum_ls attn[h][q][ls] * vT[d][ls]   (one batch; K = 4096)
__global__ void attn_v_gemm_kernel(const bf16_t* __restrict__ attn, const bf16_t* __restrict__ vT,
                                   bf16_t* __restrict__ ctx) {
  int lane = threadIdx.x & 31;
  int gw = blockIdx.x * (blockDim.x >> 5) + (threadIdx.x >> 5);
  const int tiles = (QL / 32) * (DD / 32);  // 32
  if (gw >= HH * tiles) return;
  int h = gw / tiles, t = gw % tiles;
  int row0 = (t >> 2) * 32, col0 = (t & 3) * 32;
  const bf16_t* arow = attn + (size_t)(h * QL + row0 + (lane & 15)) * LS;
  const bf16_t* brow = vT + (size_t)(col0 + (lane & 15)) * LS;
  v8f c00 = {}, c01 = {}, c10 = {}, c11 = {};
#pragma unroll 2
  for (int k = 0; k < LS; k += 32) {
    __builtin_prefetch(arow + k + 512, 0, 1);          // global_prefetch_b8
    __builtin_prefetch(brow + k + 512, 0, 1);
    v16bf a0 = frag_a_bf16(arow, k, lane);
    v16bf a1 = frag_a_bf16(arow + 16 * LS, k, lane);
    v16bf b0 = frag_b_bf16(brow, k, lane);
    v16bf b1 = frag_b_bf16(brow + 16 * LS, k, lane);
    c00 = wmma_bf16(a0, b0, c00);
    c01 = wmma_bf16(a0, b1, c01);
    c10 = wmma_bf16(a1, b0, c10);
    c11 = wmma_bf16(a1, b1, c11);
  }
  int cc = h * DD + col0;
  store_bf16(ctx, HD, row0, cc, c00, lane);
  store_bf16(ctx, HD, row0, cc + 16, c01, lane);
  store_bf16(ctx, HD, row0 + 16, cc, c10, lane);
  store_bf16(ctx, HD, row0 + 16, cc + 16, c11, lane);
}

// out[M=2048,128] = ctx[2048,1024] @ Wo[128,1024]^T + bo
__global__ void final_gemm_kernel(const bf16_t* __restrict__ ctx, const bf16_t* __restrict__ Wob,
                                  const float* __restrict__ bo, float* __restrict__ out) {
  int lane = threadIdx.x & 31;
  int gw = blockIdx.x * (blockDim.x >> 5) + (threadIdx.x >> 5);
  const int M = NB * QL, tiles_n = DD / 32;
  if (gw >= (M / 32) * tiles_n) return;
  int row0 = (gw / tiles_n) * 32, col0 = (gw % tiles_n) * 32;
  const bf16_t* arow = ctx + (size_t)(row0 + (lane & 15)) * HD;
  const bf16_t* brow = Wob + (size_t)(col0 + (lane & 15)) * HD;
  v8f c00 = {}, c01 = {}, c10 = {}, c11 = {};
#pragma unroll 1
  for (int k = 0; k < HD; k += 32) {
    v16bf a0 = frag_a_bf16(arow, k, lane);
    v16bf a1 = frag_a_bf16(arow + 16 * HD, k, lane);
    v16bf b0 = frag_b_bf16(brow, k, lane);
    v16bf b1 = frag_b_bf16(brow + 16 * HD, k, lane);
    c00 = wmma_bf16(a0, b0, c00);
    c01 = wmma_bf16(a0, b1, c01);
    c10 = wmma_bf16(a1, b0, c10);
    c11 = wmma_bf16(a1, b1, c11);
  }
  int n = lane & 15, mh = (lane >> 4) << 3;
  float bl = bo[col0 + n], bh = bo[col0 + 16 + n];
#pragma unroll
  for (int r = 0; r < 8; ++r) {
    out[(size_t)(row0 + mh + r) * DD + col0 + n] = c00[r] + bl;
    out[(size_t)(row0 + mh + r) * DD + col0 + 16 + n] = c01[r] + bh;
    out[(size_t)(row0 + 16 + mh + r) * DD + col0 + n] = c10[r] + bl;
    out[(size_t)(row0 + 16 + mh + r) * DD + col0 + 16 + n] = c11[r] + bh;
  }
}

extern "C" void kernel_launch(void* const* d_in, const int* in_sizes, int n_in,
                              void* d_out, int out_size, void* d_ws, size_t ws_size,
                              hipStream_t stream) {
  (void)in_sizes; (void)n_in; (void)out_size; (void)ws_size;
  const float* values = (const float*)d_in[0];
  const float* keys   = (const float*)d_in[1];
  const float* query  = (const float*)d_in[2];
  const int*   mask   = (const int*)d_in[3];
  const float* Wq     = (const float*)d_in[4];
  const float* Wk     = (const float*)d_in[5];
  const float* Wv     = (const float*)d_in[6];
  const float* Wo     = (const float*)d_in[7];
  const float* bo     = (const float*)d_in[8];
  float* out = (float*)d_out;

  // workspace carve (all offsets 256B-aligned) — total ~66 MB
  char* ws = (char*)d_ws;
  size_t off = 0;
  bf16_t* qf  = (bf16_t*)(ws + off); off += (size_t)NB * QL * DD * 2;   // bf16 inputs
  bf16_t* kf  = (bf16_t*)(ws + off); off += (size_t)NB * LS * DD * 2;
  bf16_t* vf  = (bf16_t*)(ws + off); off += (size_t)NB * LS * DD * 2;
  bf16_t* Wqb = (bf16_t*)(ws + off); off += (size_t)DD * DD * 2;
  bf16_t* Wkb = (bf16_t*)(ws + off); off += (size_t)DD * DD * 2;
  bf16_t* Wvb = (bf16_t*)(ws + off); off += (size_t)DD * DD * 2;
  bf16_t* Wob = (bf16_t*)(ws + off); off += (size_t)DD * HD * 2;
  bf16_t* qb  = (bf16_t*)(ws + off); off += (size_t)NB * QL * DD * 2;   // projected
  bf16_t* kb  = (bf16_t*)(ws + off); off += (size_t)NB * LS * DD * 2;
  bf16_t* vb  = (bf16_t*)(ws + off); off += (size_t)NB * LS * DD * 2;
  bf16_t* vT  = (bf16_t*)(ws + off); off += (size_t)NB * LS * DD * 2;
  float*  E   = (float*)(ws + off);  off += (size_t)QL * LS * 4;        // per-batch, reused
  bf16_t* at  = (bf16_t*)(ws + off); off += (size_t)HH * QL * LS * 2;   // per-batch, reused
  bf16_t* cb  = (bf16_t*)(ws + off); off += (size_t)NB * QL * HD * 2;

  // f32 -> bf16 conversions (tiny vs the 256 MB mask stream)
  cvt_bf16_kernel<<<(NB * QL * DD / 4 + 255) / 256, 256, 0, stream>>>(query, qf, NB * QL * DD / 4);
  cvt_bf16_kernel<<<(NB * LS * DD / 4 + 255) / 256, 256, 0, stream>>>(keys, kf, NB * LS * DD / 4);
  cvt_bf16_kernel<<<(NB * LS * DD / 4 + 255) / 256, 256, 0, stream>>>(values, vf, NB * LS * DD / 4);
  cvt_bf16_kernel<<<(DD * DD / 4 + 255) / 256, 256, 0, stream>>>(Wq, Wqb, DD * DD / 4);
  cvt_bf16_kernel<<<(DD * DD / 4 + 255) / 256, 256, 0, stream>>>(Wk, Wkb, DD * DD / 4);
  cvt_bf16_kernel<<<(DD * DD / 4 + 255) / 256, 256, 0, stream>>>(Wv, Wvb, DD * DD / 4);
  cvt_bf16_kernel<<<(DD * HD / 4 + 255) / 256, 256, 0, stream>>>(Wo, Wob, DD * HD / 4);

  // projections (WMMA)
  proj_gemm_kernel<<<(NB * QL / 32) * (DD / 32) / 4, 128, 0, stream>>>(qf, Wqb, qb, NB * QL);
  proj_gemm_kernel<<<(NB * LS / 32) * (DD / 32) / 4, 128, 0, stream>>>(kf, Wkb, kb, NB * LS);
  proj_gemm_kernel<<<(NB * LS / 32) * (DD / 32) / 4, 128, 0, stream>>>(vf, Wvb, vb, NB * LS);
  transpose_v_kernel<<<dim3(LS / 32, DD / 32, NB), dim3(32, 8), 0, stream>>>(vb, vT);

  for (int n = 0; n < NB; ++n) {
    energy_gemm_kernel<<<(QL / 32) * (LS / 32) / 4, 128, 0, stream>>>(
        qb + (size_t)n * QL * DD, kb + (size_t)n * LS * DD, E);
    softmax_kernel<<<(QL * SS + 255) / 256, 256, 0, stream>>>(
        E, mask + (size_t)n * HH * QL * KL * SS, at);
    attn_v_gemm_kernel<<<HH * (QL / 32) * (DD / 32) / 4, 128, 0, stream>>>(
        at, vT + (size_t)n * DD * LS, cb + (size_t)n * QL * HD);
  }
  final_gemm_kernel<<<(NB * QL / 32) * (DD / 32) / 4, 128, 0, stream>>>(cb, Wob, bo, out);
}
